// AdaptiveRankLinear_12086037971009
// MI455X (gfx1250) — matrix operand order repaired and checked by reading, not compiled, not run on hardware
//
#include <hip/hip_runtime.h>

// ---------------------------------------------------------------------------
// AdaptiveRankLinear, factorized low-rank path on MI455X (gfx1250):
//   t = x @ (S * Vt)^T        [16384, 256]   (bf16 intermediate, f32 accum)
//   y = t @ U^T + bias        [16384, 4096]  (f32 out)
// bf16 V_WMMA_F32_16X16X32_BF16; double-buffered LDS software pipeline;
// async global->LDS (ASYNCcnt) for bf16 tiles; native packed bf16 converts.
// ---------------------------------------------------------------------------

typedef __bf16 bf16_t;
typedef __bf16 v4bf  __attribute__((ext_vector_type(4)));
typedef __bf16 v8bf  __attribute__((ext_vector_type(8)));
typedef __bf16 v16bf __attribute__((ext_vector_type(16)));
typedef float  v4f   __attribute__((ext_vector_type(4)));
typedef float  v8f   __attribute__((ext_vector_type(8)));
typedef int    v4i   __attribute__((ext_vector_type(4)));

#define DEVFN static __device__ __forceinline__

#if defined(__HIP_DEVICE_COMPILE__) &&                                \
    __has_builtin(__builtin_amdgcn_global_load_async_to_lds_b128) && \
    __has_builtin(__builtin_amdgcn_s_wait_asynccnt)
#define USE_ASYNC_LDS 1
typedef __attribute__((address_space(1))) v4i* gptr_b128;  // global int4*
typedef __attribute__((address_space(3))) v4i* lptr_b128;  // LDS int4*
#else
#define USE_ASYNC_LDS 0
#endif

// Problem constants (reference: B=8, T=2048, IN=4096, OUT=4096, RANK=256)
constexpr int kM    = 8 * 2048;   // 16384 rows of x
constexpr int kIN   = 4096;
constexpr int kOUT  = 4096;
constexpr int kRANK = 256;

// ---------------------------------------------------------------------------
// Prep: Vt_s[r,i] = bf16(S[r] * Vt[r,i]);  U_bf[o,r] = bf16(U[o,r])
// ---------------------------------------------------------------------------
__global__ __launch_bounds__(256) void prep_vt_kernel(
    const float* __restrict__ Vt, const float* __restrict__ S,
    bf16_t* __restrict__ out) {
  int i = blockIdx.x * 256 + threadIdx.x;   // over RANK*IN
  int r = i >> 12;                          // IN = 4096
  out[i] = (bf16_t)(S[r] * Vt[i]);
}

__global__ __launch_bounds__(256) void prep_u_kernel(
    const float* __restrict__ U, bf16_t* __restrict__ out) {
  int i = blockIdx.x * 256 + threadIdx.x;   // over OUT*RANK
  out[i] = (bf16_t)U[i];
}

// Async (or sync fallback) fill of an RR x 32 bf16 tile into LDS.
template <int RR, int LDAP, int K_TOT>
DEVFN void fill_tile_bf16(bf16_t (*dst)[LDAP], const bf16_t* srcBase,
                          int row0, int k0, int tid) {
  const int r = tid >> 2;          // 64 rows / pass
  const int c = (tid & 3) * 8;
#pragma unroll
  for (int p = 0; p < RR / 64; ++p) {
    const int row = r + p * 64;
    const bf16_t* src = srcBase + (size_t)(row0 + row) * K_TOT + k0 + c;
#if USE_ASYNC_LDS
    __builtin_amdgcn_global_load_async_to_lds_b128(
        (gptr_b128)(const void*)src, (lptr_b128)(void*)&dst[row][c], 0, 0);
#else
    *(v8bf*)&dst[row][c] = *(const v8bf*)src;
#endif
  }
}

// ---------------------------------------------------------------------------
// Pipelined tiled bf16 GEMM: C[M,N_TOT] = A[M,K_TOT] * B[N_TOT,K_TOT]^T (+bias)
//   BM x BN x 32 tiles; 256 threads = 8 waves (WROWS x WCOLS); 2 LDS buffers.
// ---------------------------------------------------------------------------
template <int BM, int BN, int WROWS, int WCOLS, bool A_F32, bool OUT_F32BIAS,
          int N_STRIDE, int K_TOT>
__global__ __launch_bounds__(256) void gemm_bf16_kernel(
    const void* __restrict__ Aptr, const bf16_t* __restrict__ Bptr,
    const float* __restrict__ bias, void* __restrict__ Cptr) {
  static_assert(WROWS * WCOLS == 8, "8 waves / 256 threads");
  constexpr int BK    = 32;
  constexpr int LDA   = 40;                 // padded LDS row stride (80B)
  constexpr int MI    = BM / (WROWS * 16);
  constexpr int NI    = BN / (WCOLS * 16);
  constexpr int NIT   = K_TOT / BK;
  constexpr int APASS = BM / 32;            // f32 A-tile passes (v4f each)

  __shared__ bf16_t lA[2][BM][LDA];
  __shared__ bf16_t lB[2][BN][LDA];

  const int tid  = threadIdx.x;
  const int lane = tid & 31;
  const int w    = tid >> 5;
  const int wm   = w % WROWS;
  const int wn   = w / WROWS;
  const int half = lane >> 4;    // K-split lane half per ISA 16-bit layouts
  const int l16  = lane & 15;

  const int m0 = blockIdx.x * BM;
  const int n0 = blockIdx.y * BN;

  v8f acc[MI][NI];
#pragma unroll
  for (int i = 0; i < MI; ++i)
#pragma unroll
    for (int j = 0; j < NI; ++j) acc[i][j] = v8f{};

  // fp32 A staging registers (held across the WMMA block to hide latency)
  v4f areg[APASS];

  auto loadA_f32 = [&](int k0) {
    const float* A = (const float*)Aptr;
    const int r = tid >> 3, c = (tid & 7) * 4;
#pragma unroll
    for (int p = 0; p < APASS; ++p)
      areg[p] = *(const v4f*)(A + (size_t)(m0 + r + p * 32) * K_TOT + k0 + c);
  };
  auto storeA_f32 = [&](int buf) {
    const int r = tid >> 3, c = (tid & 7) * 4;
#pragma unroll
    for (int p = 0; p < APASS; ++p) {
      v4bf b;
      b[0] = (bf16_t)areg[p][0];
      b[1] = (bf16_t)areg[p][1];
      b[2] = (bf16_t)areg[p][2];
      b[3] = (bf16_t)areg[p][3];
      *(v4bf*)&lA[buf][r + p * 32][c] = b;   // 8B ds_store
    }
  };

  // ---- prologue: tile 0 -> buffer 0
  if (A_F32) {
    loadA_f32(0);
    storeA_f32(0);
  } else {
    fill_tile_bf16<BM, LDA, K_TOT>(lA[0], (const bf16_t*)Aptr, m0, 0, tid);
  }
  fill_tile_bf16<BN, LDA, K_TOT>(lB[0], Bptr, n0, 0, tid);
#if USE_ASYNC_LDS
  __builtin_amdgcn_s_wait_asynccnt(0);
#endif
  __syncthreads();

  for (int it = 0; it < NIT; ++it) {
    const int buf  = it & 1;
    const int nbuf = buf ^ 1;
    const int k1   = (it + 1) * BK;
    const bool haveNext = (it + 1) < NIT;

    // ---- issue next tile's fills first (overlap with WMMA below)
    if (haveNext) {
      if (A_F32) {
        loadA_f32(k1);                      // global loads; consumed post-WMMA
        if (k1 + BK < K_TOT)                // global_prefetch_b8, 2 tiles ahead
          __builtin_prefetch(
              (const char*)Aptr +
                  ((size_t)(m0 + (tid % BM)) * K_TOT + k1 + BK) * 4,
              0, 1);
      } else {
        fill_tile_bf16<BM, LDA, K_TOT>(lA[nbuf], (const bf16_t*)Aptr, m0, k1,
                                       tid);
      }
      fill_tile_bf16<BN, LDA, K_TOT>(lB[nbuf], Bptr, n0, k1, tid);
    }

    // ---- gather WMMA fragments from LDS in contiguous k order (k = 16*half + e), same for A and B
    // A 16x32: lanes<16 hold K 0..15 ; lanes>=16 hold K 16..31
    v16bf afrag[MI];
#pragma unroll
    for (int mi = 0; mi < MI; ++mi) {
      const int m = wm * (MI * 16) + mi * 16 + l16;
      v8bf x0 = *(const v8bf*)&lA[buf][m][half * 16];
      v8bf x1 = *(const v8bf*)&lA[buf][m][half * 16 + 8];
#pragma unroll
      for (int e = 0; e < 8; ++e) {
        afrag[mi][e]     = x0[e];
        afrag[mi][e + 8] = x1[e];
      }
    }
    // B 32x16: lanes<16 hold K 0..15 ; lanes>=16 hold K 16..31 (col = l16)
    v16bf bfrag[NI];
#pragma unroll
    for (int ni = 0; ni < NI; ++ni) {
      const int n = wn * (NI * 16) + ni * 16 + l16;
      v8bf x0 = *(const v8bf*)&lB[buf][n][half * 16];
      v8bf x1 = *(const v8bf*)&lB[buf][n][half * 16 + 8];
#pragma unroll
      for (int e = 0; e < 8; ++e) {
        bfrag[ni][e]     = x0[e];
        bfrag[ni][e + 8] = x1[e];
      }
    }

    // ---- MI*NI x v_wmma_f32_16x16x32_bf16
#pragma unroll
    for (int mi = 0; mi < MI; ++mi)
#pragma unroll
      for (int ni = 0; ni < NI; ++ni)
        acc[mi][ni] = __builtin_amdgcn_wmma_f32_16x16x32_bf16(
            false, afrag[mi], false, bfrag[ni], (short)0, acc[mi][ni],
            false, false);

    // ---- publish next A tile (fp32 path): convert+store after compute
    if (A_F32 && haveNext) storeA_f32(nbuf);

#if USE_ASYNC_LDS
    __builtin_amdgcn_s_wait_asynccnt(0);    // our async fills landed
#endif
    __syncthreads();  // publishes buf^1, retires all reads of buf
  }

  // ---- epilogue: C/D f32 layout: VGPR v -> M = v + 8*half ; N = l16
#pragma unroll
  for (int mi = 0; mi < MI; ++mi) {
#pragma unroll
    for (int ni = 0; ni < NI; ++ni) {
      const int gn = n0 + wn * (NI * 16) + ni * 16 + l16;
      const int gm = m0 + wm * (MI * 16) + mi * 16 + half * 8;
      if (OUT_F32BIAS) {
        const float bv = bias[gn];
        float* Cf = (float*)Cptr;
#pragma unroll
        for (int v = 0; v < 8; ++v)
          Cf[(size_t)(gm + v) * N_STRIDE + gn] = acc[mi][ni][v] + bv;
      } else {
        bf16_t* Cb = (bf16_t*)Cptr;
#pragma unroll
        for (int v = 0; v < 8; ++v)
          Cb[(size_t)(gm + v) * N_STRIDE + gn] = (bf16_t)acc[mi][ni][v];
      }
    }
  }
}

// ---------------------------------------------------------------------------
// Launch
// ---------------------------------------------------------------------------
extern "C" void kernel_launch(void* const* d_in, const int* in_sizes, int n_in,
                              void* d_out, int out_size, void* d_ws,
                              size_t ws_size, hipStream_t stream) {
  const float* x    = (const float*)d_in[0];  // [M, IN]
  const float* U    = (const float*)d_in[1];  // [OUT, RANK]
  const float* S    = (const float*)d_in[2];  // [RANK]
  const float* Vt   = (const float*)d_in[3];  // [RANK, IN]
  const float* bias = (const float*)d_in[4];  // [OUT]
  float* out = (float*)d_out;                 // [M, OUT]

  // workspace layout (bf16): Vt_s [RANK,IN] | U_bf [OUT,RANK] | t [M,RANK]
  bf16_t* vt_s = (bf16_t*)d_ws;
  bf16_t* u_bf = vt_s + (size_t)kRANK * kIN;
  bf16_t* tbuf = u_bf + (size_t)kOUT * kRANK;

  prep_vt_kernel<<<(kRANK * kIN) / 256, 256, 0, stream>>>(Vt, S, vt_s);
  prep_u_kernel<<<(kOUT * kRANK) / 256, 256, 0, stream>>>(U, u_bf);

  // GEMM1: t[M, RANK] = x[M, IN] @ Vt_s[RANK, IN]^T   (bf16 out)
  // BM=64 -> 512 blocks for latency hiding on the 256 MB x read.
  gemm_bf16_kernel<64, 128, 2, 4, true, false, kRANK, kIN>
      <<<dim3(kM / 64, kRANK / 128), 256, 0, stream>>>(
          (const void*)x, vt_s, nullptr, (void*)tbuf);

  // GEMM2: y[M, OUT] = t[M, RANK] @ U_bf[OUT, RANK]^T + bias  (f32 out)
  gemm_bf16_kernel<128, 128, 4, 2, false, true, kOUT, kRANK>
      <<<dim3(kM / 128, kOUT / 128), 256, 0, stream>>>(
          (const void*)tbuf, u_bf, bias, (void*)out);
}
